// GATv2Conv_Model_84499186582058
// MI455X (gfx1250) — matrix operand order repaired and checked by reading, not compiled, not run on hardware
//
#include <hip/hip_runtime.h>
#include <hip/hip_bf16.h>

typedef __attribute__((ext_vector_type(16))) __bf16 v16bf;
typedef __attribute__((ext_vector_type(8)))  __bf16 v8bf;
typedef __attribute__((ext_vector_type(8)))  float  v8f;

#define IN_C    116
#define OUT_DIM 64
#define HEADS   4
#define OUT_C   16
#define NGRAPH  100
#define NEG_SLOPE 0.2f
#define BN_EPS  1e-5f

// ---------------- helpers: monotone float<->uint key for atomic max ---------
__device__ __forceinline__ unsigned f2key(float f) {
    unsigned u = __float_as_uint(f);
    return (u & 0x80000000u) ? ~u : (u | 0x80000000u);
}
__device__ __forceinline__ float key2f(unsigned k) {
    unsigned u = (k & 0x80000000u) ? (k & 0x7FFFFFFFu) : ~k;
    return __uint_as_float(u);
}

// ---------------- zero scratch accumulators ---------------------------------
__global__ void zero_kernel(unsigned* p, size_t n) {
    size_t i = (size_t)blockIdx.x * blockDim.x + threadIdx.x;
    size_t stride = (size_t)gridDim.x * blockDim.x;
    for (; i < n; i += stride) p[i] = 0u;
}

// ---------------- fused 3-GEMM: xl = x@Wl+bl, xr = x@Wr+br, xres = x@Wres+bres
// block = 128 threads (4 waves). Block handles 64 rows; wave w does rows
// [64*blk + 16*w, +16) x 64 cols with v_wmma_f32_16x16x32_bf16, K padded 116->128.
// Weights are staged in LDS pre-swizzled into per-lane WMMA B-fragment order so
// each fragment load is one contiguous 32B chunk (2x ds_load_b128, no packing).
__global__ void gemm3_kernel(const float* __restrict__ x,
                             const float* __restrict__ Wl, const float* __restrict__ bl,
                             const float* __restrict__ Wr, const float* __restrict__ br,
                             const float* __restrict__ Wres, const float* __restrict__ bres,
                             float* __restrict__ xl, float* __restrict__ xr,
                             float* __restrict__ xres, int N)
{
    __shared__ __bf16 Abuf[64][128];               // 16 KB, row-major x tile
    __shared__ __bf16 Wfrag[48 * 32 * 16];         // 48 KB, fragment-order weights
    // frag index = (t*4 + ct)*4 + ks ; element = frag*512 + lane*16 + e

    const int tid = threadIdx.x;
    const int rowbase = blockIdx.x * 64;

    // stage weights, swizzled into WMMA B-fragment layout:
    // B 32x16 bf16, contiguous k order used here (same as A): lane = khalf*16 + (col%16); element e = k%16; khalf = (k%32)/16
    const float* Ws[3] = {Wl, Wr, Wres};
    #pragma unroll
    for (int t = 0; t < 3; ++t) {
        for (int i = tid; i < 128 * 64; i += 128) {
            int k = i >> 6, c = i & 63;
            float v = (k < IN_C) ? Ws[t][k * OUT_DIM + c] : 0.0f;
            int ks = k >> 5;
            int kl = k & 31;
            int khalf = kl >> 4;
            int e  = kl & 15;
            int lane = khalf * 16 + (c & 15);
            int ct = c >> 4;
            int frag = (t * 4 + ct) * 4 + ks;
            Wfrag[(frag * 32 + lane) * 16 + e] = (__bf16)v;
        }
    }
    // stage x tile (zero-padded K 116->128, rows beyond N -> 0)
    for (int i = tid; i < 64 * 128; i += 128) {
        int r = i >> 7, k = i & 127;
        int gr = rowbase + r;
        float v = (gr < N && k < IN_C) ? x[(size_t)gr * IN_C + k] : 0.0f;
        Abuf[r][k] = (__bf16)v;
    }
    __syncthreads();

    const int wave  = tid >> 5;
    const int lane  = tid & 31;
    const int mlane = lane & 15;
    const int khalf = lane >> 4;
    const int arow  = wave * 16 + mlane;

    // hoist the 4 A fragments (reused across all 12 output tiles).
    // A 16x32 bf16, contiguous k order used here (same as B): lane m holds K = khalf*16 + {0..15}
    v16bf afrag[4];
    #pragma unroll
    for (int ks = 0; ks < 4; ++ks) {
        afrag[ks] = *(const v16bf*)&Abuf[arow][ks * 32 + khalf * 16];
    }

    float* outs[3] = {xl, xr, xres};
    const float* biases[3] = {bl, br, bres};

    #pragma unroll
    for (int t = 0; t < 3; ++t) {
        #pragma unroll
        for (int ct = 0; ct < 4; ++ct) {
            v8f acc = {};
            #pragma unroll
            for (int ks = 0; ks < 4; ++ks) {
                int frag = (t * 4 + ct) * 4 + ks;
                const v16bf b = *(const v16bf*)&Wfrag[(frag * 32 + lane) * 16];
                acc = __builtin_amdgcn_wmma_f32_16x16x32_bf16(
                    false, afrag[ks], false, b, (short)0, acc, false, false);
            }
            // C/D layout: VGPR r -> M = r + 8*(lane>=16), N = lane&15
            int n  = ct * 16 + mlane;
            int mb = khalf * 8;
            float bias = biases[t][n];
            #pragma unroll
            for (int r = 0; r < 8; ++r) {
                int grow = rowbase + wave * 16 + mb + r;
                if (grow < N) outs[t][(size_t)grow * OUT_DIM + n] = acc[r] + bias;
            }
        }
    }
}

// ---------------- edge scores + segment max ---------------------------------
__global__ void edge_logits_kernel(const float* __restrict__ xl, const float* __restrict__ xr,
                                   const int* __restrict__ ei, const float* __restrict__ ea,
                                   const float* __restrict__ We, const float* __restrict__ att,
                                   float* __restrict__ logits, unsigned* __restrict__ nmax,
                                   int E, int N)
{
    int e = blockIdx.x * blockDim.x + threadIdx.x;
    if (e >= E + N) return;
    int s, d; float a;
    if (e < E) { s = ei[e]; d = ei[E + e]; a = ea[e]; }
    else       { s = d = e - E; a = 1.0f; }

    const float4* pl = (const float4*)(xl + (size_t)s * OUT_DIM);
    const float4* pr = (const float4*)(xr + (size_t)d * OUT_DIM);
    const float4* pw = (const float4*)We;
    const float4* pa = (const float4*)att;

    #pragma unroll
    for (int h = 0; h < HEADS; ++h) {
        float logit = 0.0f;
        #pragma unroll
        for (int q = 0; q < 4; ++q) {
            float4 vl = pl[h * 4 + q];
            float4 vr = pr[h * 4 + q];
            float4 we = pw[h * 4 + q];
            float4 at = pa[h * 4 + q];
            float t0 = vl.x + vr.x + a * we.x; t0 = t0 > 0.f ? t0 : NEG_SLOPE * t0;
            float t1 = vl.y + vr.y + a * we.y; t1 = t1 > 0.f ? t1 : NEG_SLOPE * t1;
            float t2 = vl.z + vr.z + a * we.z; t2 = t2 > 0.f ? t2 : NEG_SLOPE * t2;
            float t3 = vl.w + vr.w + a * we.w; t3 = t3 > 0.f ? t3 : NEG_SLOPE * t3;
            logit += at.x * t0 + at.y * t1 + at.z * t2 + at.w * t3;
        }
        logits[(size_t)e * HEADS + h] = logit;
        atomicMax(&nmax[(size_t)d * HEADS + h], f2key(logit));
    }
}

// ---------------- exp(logit - max) and segment sum --------------------------
__global__ void edge_ex_kernel(const int* __restrict__ ei,
                               float* __restrict__ logits,
                               const unsigned* __restrict__ nmax,
                               float* __restrict__ denom, int E, int N)
{
    int e = blockIdx.x * blockDim.x + threadIdx.x;
    if (e >= E + N) return;
    int d = (e < E) ? ei[E + e] : (e - E);
    #pragma unroll
    for (int h = 0; h < HEADS; ++h) {
        float m  = key2f(nmax[(size_t)d * HEADS + h]);
        float ex = __expf(logits[(size_t)e * HEADS + h] - m);
        logits[(size_t)e * HEADS + h] = ex;
        atomicAdd(&denom[(size_t)d * HEADS + h], ex);
    }
}

// ---------------- alpha-weighted scatter of source messages -----------------
__global__ void edge_agg_kernel(const int* __restrict__ ei,
                                const float* __restrict__ ex,
                                const float* __restrict__ denom,
                                const float* __restrict__ xl,
                                float* __restrict__ hbuf, int E, int N)
{
    long long t = (long long)blockIdx.x * blockDim.x + threadIdx.x;
    long long total = (long long)(E + N) * HEADS;
    if (t >= total) return;
    int e = (int)(t >> 2);
    int h = (int)(t & 3);
    int s, d;
    if (e < E) { s = ei[e]; d = ei[E + e]; }
    else       { s = d = e - E; }
    float alpha = ex[(size_t)e * HEADS + h] / denom[(size_t)d * HEADS + h];
    const float* src = xl + (size_t)s * OUT_DIM + h * OUT_C;
    float* dst = hbuf + (size_t)d * OUT_DIM + h * OUT_C;
    #pragma unroll
    for (int c = 0; c < OUT_C; ++c)
        atomicAdd(&dst[c], alpha * src[c]);
}

// ---------------- add gat bias + BN statistics (LDS-reduced) ----------------
__global__ void bn_stats_kernel(float* __restrict__ hbuf,
                                const float* __restrict__ gat_bias,
                                float* __restrict__ bnsum, float* __restrict__ bnsumsq,
                                int N)
{
    __shared__ float rs[4][64];
    __shared__ float rq[4][64];
    int tid = threadIdx.x;            // 256 threads
    int c   = tid & 63;
    int grp = tid >> 6;               // 0..3
    int base = blockIdx.x * 256;
    float bias = gat_bias[c];
    float s = 0.f, q = 0.f;
    for (int i = 0; i < 64; ++i) {
        int n = base + grp + 4 * i;
        if (n < N) {
            float v = hbuf[(size_t)n * OUT_DIM + c] + bias;
            hbuf[(size_t)n * OUT_DIM + c] = v;
            s += v; q += v * v;
        }
    }
    rs[grp][c] = s; rq[grp][c] = q;
    __syncthreads();
    if (grp == 0) {
        float ts = rs[0][c] + rs[1][c] + rs[2][c] + rs[3][c];
        float tq = rq[0][c] + rq[1][c] + rq[2][c] + rq[3][c];
        atomicAdd(&bnsum[c], ts);
        atomicAdd(&bnsumsq[c], tq);
    }
}

// ---------------- BN apply + residual + per-graph pooling -------------------
__global__ void bn_apply_pool_kernel(const float* __restrict__ hbuf,
                                     const float* __restrict__ xres,
                                     const float* __restrict__ bnsum,
                                     const float* __restrict__ bnsumsq,
                                     const float* __restrict__ gamma,
                                     const float* __restrict__ beta,
                                     const int* __restrict__ batch,
                                     float* __restrict__ pool, float* __restrict__ cnt,
                                     int N)
{
    long long t = (long long)blockIdx.x * blockDim.x + threadIdx.x;
    if (t >= (long long)N * OUT_DIM) return;
    int n = (int)(t >> 6);
    int c = (int)(t & 63);
    float mu  = bnsum[c] / (float)N;
    float var = bnsumsq[c] / (float)N - mu * mu;
    float h = hbuf[(size_t)n * OUT_DIM + c];
    float v = gamma[c] * (h - mu) * rsqrtf(var + BN_EPS) + beta[c]
            + xres[(size_t)n * OUT_DIM + c];
    int g = batch[n];
    atomicAdd(&pool[(size_t)g * OUT_DIM + c], v);
    if (c == 0) atomicAdd(&cnt[g], 1.0f);
}

// ---------------- tiny MLP head: [100,64] -> relu 64 -> 2 -------------------
__global__ void mlp_kernel(const float* __restrict__ pool, const float* __restrict__ cnt,
                           const float* __restrict__ W1, const float* __restrict__ b1,
                           const float* __restrict__ W2, const float* __restrict__ b2,
                           float* __restrict__ out)
{
    __shared__ float pl[64];
    __shared__ float z[64];
    int g = blockIdx.x;
    int c = threadIdx.x;      // 64 threads
    float cc = cnt[g]; cc = cc > 1.0f ? cc : 1.0f;
    pl[c] = pool[(size_t)g * OUT_DIM + c] / cc;
    __syncthreads();
    float acc = b1[c];
    #pragma unroll 8
    for (int k = 0; k < 64; ++k) acc += pl[k] * W1[k * 64 + c];
    z[c] = acc > 0.f ? acc : 0.f;
    __syncthreads();
    if (c < 2) {
        float o = b2[c];
        for (int k = 0; k < 64; ++k) o += z[k] * W2[k * 2 + c];
        out[g * 2 + c] = o;
    }
}

extern "C" void kernel_launch(void* const* d_in, const int* in_sizes, int n_in,
                              void* d_out, int out_size, void* d_ws, size_t ws_size,
                              hipStream_t stream) {
    const float* x        = (const float*)d_in[0];
    const int*   ei       = (const int*)  d_in[1];
    const float* ea       = (const float*)d_in[2];
    const int*   batch    = (const int*)  d_in[3];
    const float* Wl       = (const float*)d_in[4];
    const float* bl       = (const float*)d_in[5];
    const float* Wr       = (const float*)d_in[6];
    const float* br       = (const float*)d_in[7];
    const float* We       = (const float*)d_in[8];
    const float* att      = (const float*)d_in[9];
    const float* gat_bias = (const float*)d_in[10];
    const float* gamma    = (const float*)d_in[11];
    const float* beta     = (const float*)d_in[12];
    const float* Wres     = (const float*)d_in[13];
    const float* bres     = (const float*)d_in[14];
    const float* W1       = (const float*)d_in[15];
    const float* b1       = (const float*)d_in[16];
    const float* W2       = (const float*)d_in[17];
    const float* b2       = (const float*)d_in[18];
    float* out = (float*)d_out;

    const int N = in_sizes[0] / IN_C;     // 50000
    const int E = in_sizes[1] / 2;        // 800000
    const int EN = E + N;

    // workspace layout (floats)
    float* ws   = (float*)d_ws;
    size_t No64 = (size_t)N * OUT_DIM;
    float* xl     = ws;
    float* xr     = ws + No64;
    float* xres   = ws + 2 * No64;
    float* logits = ws + 3 * No64;                       // (E+N)*4, reused as ex
    unsigned* nmax = (unsigned*)(logits + (size_t)EN * HEADS);
    float* denom  = (float*)(nmax + (size_t)N * HEADS);
    float* hbuf   = denom + (size_t)N * HEADS;
    float* bnsum  = hbuf + No64;
    float* bnsumsq = bnsum + 64;
    float* pool   = bnsumsq + 64;
    float* cnt    = pool + (size_t)NGRAPH * OUT_DIM;

    // zero all accumulator state (nmax..cnt is contiguous)
    size_t zero_words = (size_t)N * HEADS * 2 + No64 + 64 + 64
                      + (size_t)NGRAPH * OUT_DIM + NGRAPH;
    zero_kernel<<<1024, 256, 0, stream>>>(nmax, zero_words);

    // fused WMMA GEMMs
    gemm3_kernel<<<(N + 63) / 64, 128, 0, stream>>>(
        x, Wl, bl, Wr, br, Wres, bres, xl, xr, xres, N);

    // edge phase
    int ethreads = 256;
    edge_logits_kernel<<<(EN + ethreads - 1) / ethreads, ethreads, 0, stream>>>(
        xl, xr, ei, ea, We, att, logits, nmax, E, N);
    edge_ex_kernel<<<(EN + ethreads - 1) / ethreads, ethreads, 0, stream>>>(
        ei, logits, nmax, denom, E, N);
    long long aggT = (long long)EN * HEADS;
    edge_agg_kernel<<<(int)((aggT + ethreads - 1) / ethreads), ethreads, 0, stream>>>(
        ei, logits, denom, xl, hbuf, E, N);

    // batchnorm + residual + pool
    bn_stats_kernel<<<(N + 255) / 256, 256, 0, stream>>>(hbuf, gat_bias, bnsum, bnsumsq, N);
    long long bpT = (long long)N * OUT_DIM;
    bn_apply_pool_kernel<<<(int)((bpT + 255) / 256), 256, 0, stream>>>(
        hbuf, xres, bnsum, bnsumsq, gamma, beta, batch, pool, cnt, N);

    // head
    mlp_kernel<<<NGRAPH, 64, 0, stream>>>(pool, cnt, W1, b1, W2, b2, out);
}
